// Quantize_56453050138684
// MI455X (gfx1250) — matrix operand order repaired and checked by reading, not compiled, not run on hardware
//
#include <hip/hip_runtime.h>
#include <hip/hip_bf16.h>

#define DD 256
#define KK 1024
#define NN 65536          // 32 * 2048 tokens
#define XP 264            // padded LDS row pitch (bf16 elems): 256 + 8 to spread banks

typedef __bf16 bf16x16 __attribute__((ext_vector_type(16)));
typedef float  v8f     __attribute__((ext_vector_type(8)));
typedef unsigned int v4u __attribute__((ext_vector_type(4)));
typedef int          v8i __attribute__((ext_vector_type(8)));
typedef int          v4i __attribute__((ext_vector_type(4)));

union ABPack { uint4 q[2]; bf16x16 v; };

// output layout (floats, concatenated in reference return order)
#define OFF_LOSS 0
#define OFF_Q    1
#define OFF_CL   (1 + 16777216)
#define OFF_DW   (OFF_CL + KK)
#define OFF_IDX  (OFF_DW + DD * KK)
#define OFF_PP   (OFF_IDX + NN)

__device__ __forceinline__ unsigned short f2bf(float f) {
    unsigned int u = __float_as_uint(f);
    u += 0x7fffu + ((u >> 16) & 1u);      // round-to-nearest-even
    return (unsigned short)(u >> 16);
}

// ---- TDM: DMA one 16x256 bf16 tile (row stride 256) into LDS with +4-DWORD row pad ----
__device__ __forceinline__ void tdm_load_tile(const unsigned short* gsrc, unsigned int ldsOff) {
    unsigned long long ga = (unsigned long long)(size_t)gsrc;
    v4u g0;
    g0[0] = 1u;                                             // count=1, user descriptor
    g0[1] = ldsOff;                                         // lds_addr (bytes)
    g0[2] = (unsigned int)ga;                               // global_addr[31:0]
    g0[3] = (unsigned int)((ga >> 32) & 0x1FFFFFFu)         // global_addr[56:32]
          | (2u << 30);                                     // type = 2 ("image")
    v8i g1;
    g1[0] = (1 << 16)        // data_size = 2 bytes
          | (1 << 20)        // pad_enable
          | (6 << 22)        // pad_interval: 128 DWORDs (= one 256-elem bf16 row)
          | (3 << 25);       // pad_amount: 4 DWORDs (= 8 bf16) -> 264-elem pitch
    g1[1] = (int)(256u << 16);   // tensor_dim0 = 256 (bits [79:48] low half)
    g1[2] = (int)(1024u << 16);  // tensor_dim0 hi = 0 | tensor_dim1 = 1024 (low half)
    g1[3] = (int)(256u << 16);   // tensor_dim1 hi = 0 | tile_dim0 = 256
    g1[4] = 16;                  // tile_dim1 = 16, tile_dim2 = 0
    g1[5] = 256;                 // tensor_dim0_stride[31:0] = 256
    g1[6] = 0;                   // stride hi | tensor_dim1_stride lo
    g1[7] = 0;
    v4i gz = {0, 0, 0, 0};
#if defined(__clang_major__) && __clang_major__ >= 23
    v8i gz8 = {0, 0, 0, 0, 0, 0, 0, 0};
    __builtin_amdgcn_tensor_load_to_lds(g0, g1, gz, gz, gz8, 0);
#else
    __builtin_amdgcn_tensor_load_to_lds(g0, g1, gz, gz, 0);
#endif
}

// ---- zero the atomically-accumulated regions ----
__global__ void vq_zero(float* dwws, float* cluster, float* loss) {
    int i = blockIdx.x * 256 + threadIdx.x;
    if (i < KK * DD) dwws[i] = 0.0f;
    if (i < KK)      cluster[i] = 0.0f;
    if (i == 0)      *loss = 0.0f;
}

// ---- transpose codebook: wT[k*D+d] = w[d*K+k]  (fp32 + bf16 copies) ----
__global__ void vq_prep(const float* __restrict__ w, float* __restrict__ wT,
                        unsigned short* __restrict__ wbt) {
    int i = blockIdx.x * 256 + threadIdx.x;      // i = k*256 + d
    if (i >= KK * DD) return;
    int k = i >> 8, d = i & 255;
    float v = w[d * KK + k];
    wT[i]  = v;
    wbt[i] = f2bf(v);
}

// ---- 0.5 * ||w_k||^2 ----
__global__ void vq_wsq(const float* __restrict__ w, float* __restrict__ wsqh) {
    int k = blockIdx.x * 256 + threadIdx.x;
    if (k >= KK) return;
    float s = 0.0f;
    for (int d = 0; d < DD; ++d) { float v = w[d * KK + k]; s += v * v; }
    wsqh[k] = 0.5f * s;
}

// ---- main: TDM-fed WMMA distances + argmax + gather + stats ----
__global__ __launch_bounds__(128)
void vq_main(const float* __restrict__ x, const float* __restrict__ wT,
             const unsigned short* __restrict__ wbt, const float* __restrict__ wsqh,
             float* __restrict__ dwws, float* __restrict__ qOut,
             float* __restrict__ idxOut, float* __restrict__ cluster,
             float* __restrict__ loss) {
    __shared__ __align__(16) unsigned short sX[64 * XP];      // 64 tokens x 256 d (bf16)
    __shared__ __align__(16) unsigned short sW[2][16 * XP];   // double-buffered B tiles
    __shared__ float swsq[KK];
    __shared__ int   sIdx[64];
    __shared__ float sRed[128];

    const int tid  = threadIdx.x;
    const int lane = tid & 31;
    const int wave = tid >> 5;          // 4 waves, 16 tokens each
    const int tokBase = blockIdx.x * 64;
    const int NT = KK / 16;             // 64 k-tiles

    // kick off TDM for tile 0 immediately (overlaps with X staging below)
    if (wave == 0)
        tdm_load_tile(wbt, (unsigned int)(unsigned long long)(size_t)&sW[0][0]);

    // stage X tile -> bf16 LDS (coalesced f32 reads)
    for (int i = tid; i < 64 * DD; i += 128) {
        int r = i >> 8, d = i & 255;
        sX[r * XP + d] = f2bf(x[(size_t)(tokBase + r) * DD + d]);
    }
    for (int i = tid; i < KK; i += 128) swsq[i] = wsqh[i];

    const int nl = lane & 15;           // column (N) / row (M) within half
    const int h  = lane >> 4;           // wave half
    // A layout (16-bit, 16x32): runs k = 8h..8h+7 (elems 0-7) and 16+8h.. (elems 8-15)
    const unsigned short* aBase = &sX[(wave * 16 + nl) * XP + h * 8];

    float best[8];
    int   bidx[8];
#pragma unroll
    for (int i = 0; i < 8; ++i) { best[i] = -3.4e38f; bidx[i] = 0; }

    for (int kt = 0; kt < NT; ++kt) {
        __syncthreads();               // everyone done reading buf[(kt+1)&1] (next buffer)
        if (wave == 0) {
            if (kt + 1 < NT) {
                tdm_load_tile(wbt + (size_t)(kt + 1) * 16 * DD,
                              (unsigned int)(unsigned long long)(size_t)&sW[(kt + 1) & 1][0]);
                __builtin_amdgcn_s_wait_tensorcnt(1);   // tile kt resident (in-order TDM)
            } else {
                __builtin_amdgcn_s_wait_tensorcnt(0);
            }
        }
        __syncthreads();               // buf[kt&1] visible to all waves

        // B layout (16-bit, 32x16): runs k = 8h..8h+7 (elems 0-7) and 16+8h.. (elems 8-15)
        const unsigned short* bBase = &sW[kt & 1][nl * XP + h * 8];

        v8f acc = {};
#pragma unroll
        for (int d0 = 0; d0 < DD; d0 += 32) {
            ABPack a, b;
            a.q[0] = *(const uint4*)(aBase + d0);         // k = d0+8h .. +7
            a.q[1] = *(const uint4*)(aBase + d0 + 16);    // k = d0+16+8h .. +7
            b.q[0] = *(const uint4*)(bBase + d0);         // k = d0+8h .. +7
            b.q[1] = *(const uint4*)(bBase + d0 + 16);    // k = d0+16+8h .. +7
            acc = __builtin_amdgcn_wmma_f32_16x16x32_bf16(
                false, a.v, false, b.v, (short)0, acc, false, false);
        }

        const int kcol = kt * 16 + nl;
        const float wq = swsq[kcol];
#pragma unroll
        for (int i = 0; i < 8; ++i) {
            float s = acc[i] - wq;                     // x.w - 0.5||w||^2
            if (s > best[i]) { best[i] = s; bidx[i] = kcol; }
        }
    }

    // cross-lane argmax over the 16 columns (stays inside each wave half)
#pragma unroll
    for (int i = 0; i < 8; ++i) {
        float bv = best[i];
        int   bi = bidx[i];
#pragma unroll
        for (int off = 8; off; off >>= 1) {
            float ov = __shfl_xor(bv, off, 32);
            int   oi = __shfl_xor(bi, off, 32);
            if (ov > bv || (ov == bv && oi < bi)) { bv = ov; bi = oi; }
        }
        if (nl == 0) {
            int tl = wave * 16 + i + 8 * h;            // C VGPR i <-> row i + 8h
            sIdx[tl] = bi;
            idxOut[tokBase + tl] = (float)bi;
            atomicAdd(&cluster[bi], 1.0f);
        }
    }
    __syncthreads();

    // exact fp32 gather + commitment loss + dw accumulation
    float lsum = 0.0f;
    for (int t = 0; t < 64; ++t) {
        const int kk = sIdx[t];
        const size_t tb = (size_t)(tokBase + t) * DD;
        for (int d = tid; d < DD; d += 128) {
            float q  = wT[kk * DD + d];
            float xv = x[tb + d];
            qOut[tb + d] = q;                          // straight-through forward == q
            float df = q - xv;
            lsum += df * df;
            atomicAdd(&dwws[kk * DD + d], xv);
        }
    }
    sRed[tid] = lsum;
    __syncthreads();
    for (int s = 64; s; s >>= 1) {
        if (tid < s) sRed[tid] += sRed[tid + s];
        __syncthreads();
    }
    if (tid == 0) atomicAdd(loss, sRed[0] * (1.0f / ((float)NN * (float)DD)));
}

// ---- dw transpose: dw[d*K+k] = dwws[k*D+d] ----
__global__ void vq_dwt(const float* __restrict__ dwws, float* __restrict__ dw) {
    int j = blockIdx.x * 256 + threadIdx.x;
    if (j >= KK * DD) return;
    int d = j >> 10, k = j & 1023;
    dw[j] = dwws[k * DD + d];
}

// ---- perplexity ----
__global__ void vq_perp(const float* __restrict__ cluster, float* __restrict__ perp) {
    __shared__ float red[256];
    int tid = threadIdx.x;
    float s = 0.0f;
    for (int k = tid; k < KK; k += 256) {
        float p = cluster[k] * (1.0f / (float)NN);
        s += p * __logf(p + 1e-10f);
    }
    red[tid] = s;
    __syncthreads();
    for (int st = 128; st; st >>= 1) {
        if (tid < st) red[tid] += red[tid + st];
        __syncthreads();
    }
    if (tid == 0) *perp = __expf(-red[0]);
}

extern "C" void kernel_launch(void* const* d_in, const int* in_sizes, int n_in,
                              void* d_out, int out_size, void* d_ws, size_t ws_size,
                              hipStream_t stream) {
    const float* x = (const float*)d_in[0];
    const float* w = (const float*)d_in[1];
    float* out     = (float*)d_out;

    float* loss    = out + OFF_LOSS;
    float* qOut    = out + OFF_Q;
    float* cluster = out + OFF_CL;
    float* dw      = out + OFF_DW;
    float* idxOut  = out + OFF_IDX;
    float* perp    = out + OFF_PP;

    char* ws = (char*)d_ws;
    float*          dwws = (float*)ws;                        // 1 MB
    float*          wT   = (float*)(ws + (1u << 20));         // 1 MB
    float*          wsqh = (float*)(ws + (2u << 20));         // 4 KB
    unsigned short* wbt  = (unsigned short*)(ws + (2u << 20) + 4096); // 512 KB

    vq_zero<<<(KK * DD + 255) / 256, 256, 0, stream>>>(dwws, cluster, loss);
    vq_prep<<<(KK * DD + 255) / 256, 256, 0, stream>>>(w, wT, wbt);
    vq_wsq<<<(KK + 255) / 256, 256, 0, stream>>>(w, wsqh);
    vq_main<<<NN / 64, 128, 0, stream>>>(x, wT, wbt, wsqh, dwws,
                                         qOut, idxOut, cluster, loss);
    vq_dwt<<<(KK * DD + 255) / 256, 256, 0, stream>>>(dwws, dw);
    vq_perp<<<1, 256, 0, stream>>>(cluster, perp);
}
